// FAVORPlusAttention_71794673319959
// MI455X (gfx1250) — matrix operand order repaired and checked by reading, not compiled, not run on hardware
//
#include <hip/hip_runtime.h>

typedef __attribute__((ext_vector_type(16))) __bf16 v16bf;
typedef __attribute__((ext_vector_type(8)))  float  v8f;
typedef int v4i_vs __attribute__((vector_size(16)));   // matches builtin param type
typedef __attribute__((address_space(1))) v4i_vs* gptr_v4i;
typedef __attribute__((address_space(3))) v4i_vs* lptr_v4i;

#if defined(__has_builtin)
#if __has_builtin(__builtin_amdgcn_global_load_async_to_lds_b128) && \
    __has_builtin(__builtin_amdgcn_s_wait_asynccnt)
#define USE_ASYNC 1
#else
#define USE_ASYNC 0
#endif
#else
#define USE_ASYNC 0
#endif

__device__ __forceinline__ v8f v8f_zero() {
    v8f z;
#pragma unroll
    for (int i = 0; i < 8; i++) z[i] = 0.f;
    return z;
}

// ---------------------------------------------------------------- converts
__global__ void cvt_f32_bf16(const float* __restrict__ in, __bf16* __restrict__ out, int n) {
    int i = blockIdx.x * blockDim.x + threadIdx.x;
    if (i < n) out[i] = (__bf16)in[i];
}

// -------- async (or sync fallback) stage of one 128x64 bf16 tile pair ------
__device__ __forceinline__ void stage_tile(
    __bf16 (&As)[2][128][64], __bf16 (&Bs)[2][128][64], int nb,
    const __bf16* __restrict__ Ag, const __bf16* __restrict__ Bg,
    size_t aBase, size_t bBase, int K, int k0, int tid)
{
#pragma unroll
    for (int i = 0; i < 4; i++) {
        int c   = tid + i * 256;       // 0..1023
        int row = c >> 3;
        int kc  = (c & 7) * 8;         // 8 bf16 = 16B chunks
        const __bf16* ga = Ag + aBase + (size_t)row * K + k0 + kc;
        const __bf16* gb = Bg + bBase + (size_t)row * K + k0 + kc;
#if USE_ASYNC
        __builtin_amdgcn_global_load_async_to_lds_b128(
            (gptr_v4i)(void*)ga, (lptr_v4i)(void*)&As[nb][row][kc], 0, 0);
        __builtin_amdgcn_global_load_async_to_lds_b128(
            (gptr_v4i)(void*)gb, (lptr_v4i)(void*)&Bs[nb][row][kc], 0, 0);
#else
        *(uint4*)&As[nb][row][kc] = *(const uint4*)ga;
        *(uint4*)&Bs[nb][row][kc] = *(const uint4*)gb;
#endif
    }
}

// ---------------------------------------------------------------- big GEMM
// C[M,Nc] = A[M,K](bf16) * B[Nc,K]^T(bf16) + bias  (TN, K-major both sides)
// Double-buffered LDS, 64-wide K stages, async global->LDS when available.
// MODE 0: scatter-store bf16 into qkv layout [3][B=4][H=16][N=4096][64]
// MODE 1: store f32 row-major [M,Nc] to outp
template <int MODE>
__global__ __launch_bounds__(256) void gemm_bf16_tn(
    const __bf16* __restrict__ Ag, const __bf16* __restrict__ Bg,
    const float* __restrict__ bias, float* __restrict__ outp,
    int M, int Nc, int K, __bf16* __restrict__ qkv_out)
{
    __shared__ __align__(32) __bf16 As[2][128][64];
    __shared__ __align__(32) __bf16 Bs[2][128][64];

    const int tid  = threadIdx.x;
    const int lane = tid & 31;
    const int wave = tid >> 5;          // 0..7
    const int half = lane >> 4;
    const int mrow = lane & 15;
    const int col  = lane & 15;
    const int waveM = wave >> 1;        // 0..3 : 32 rows each
    const int waveN = wave & 1;         // 0..1 : 64 cols each
    const int mBlock = blockIdx.x * 128;
    const int nBlock = blockIdx.y * 128;
    const size_t aBase = (size_t)mBlock * K;
    const size_t bBase = (size_t)nBlock * K;

    v8f acc[2][4];
#pragma unroll
    for (int i = 0; i < 2; i++)
#pragma unroll
        for (int j = 0; j < 4; j++) acc[i][j] = v8f_zero();

    const int NSTAGE = K >> 6;          // K/64
    stage_tile(As, Bs, 0, Ag, Bg, aBase, bBase, K, 0, tid);

    for (int s = 0; s < NSTAGE; s++) {
        const int buf = s & 1;
        if (s + 1 < NSTAGE)
            stage_tile(As, Bs, buf ^ 1, Ag, Bg, aBase, bBase, K, (s + 1) * 64, tid);
#if USE_ASYNC
        if (s + 1 < NSTAGE) __builtin_amdgcn_s_wait_asynccnt(8);
        else                __builtin_amdgcn_s_wait_asynccnt(0);
#endif
        __syncthreads();

#pragma unroll
        for (int ks = 0; ks < 2; ks++) {
            v16bf a[2], b[4];
#pragma unroll
            for (int mt = 0; mt < 2; mt++) {
                int rA = waveM * 32 + mt * 16 + mrow;
#pragma unroll
                for (int e = 0; e < 8; e++) a[mt][e]     = As[buf][rA][ks * 32 + half * 16 + e];
#pragma unroll
                for (int e = 0; e < 8; e++) a[mt][8 + e] = As[buf][rA][ks * 32 + half * 16 + 8 + e];
            }
#pragma unroll
            for (int nt = 0; nt < 4; nt++) {
                int rB = waveN * 64 + nt * 16 + col;
                b[nt] = *(const v16bf*)&Bs[buf][rB][ks * 32 + half * 16];
            }
#pragma unroll
            for (int mt = 0; mt < 2; mt++)
#pragma unroll
                for (int nt = 0; nt < 4; nt++)
                    acc[mt][nt] = __builtin_amdgcn_wmma_f32_16x16x32_bf16(
                        false, a[mt], false, b[nt], (short)0, acc[mt][nt], false, false);
        }
        __syncthreads();
    }

    const int cBase = nBlock + waveN * 64;
    const int mBase = mBlock + waveM * 32;
#pragma unroll
    for (int nt = 0; nt < 4; nt++) {
        int c   = cBase + nt * 16 + col;
        float bi = bias[c];
        if (MODE == 0) {
            int which = c >> 10;
            int d     = c & 1023;
            int h     = d >> 6;
            int hd_i  = d & 63;
            __bf16* base = qkv_out + (size_t)which * (size_t)4194304;
#pragma unroll
            for (int mt = 0; mt < 2; mt++)
#pragma unroll
                for (int r = 0; r < 8; r++) {
                    int m   = mBase + mt * 16 + r + 8 * half;
                    int b_i = m >> 12, n_i = m & 4095;
                    float v = acc[mt][nt][r] + bi;
                    base[(((size_t)(b_i * 16 + h) * 4096 + n_i) << 6) + hd_i] = (__bf16)v;
                }
        } else {
#pragma unroll
            for (int mt = 0; mt < 2; mt++)
#pragma unroll
                for (int r = 0; r < 8; r++) {
                    int m = mBase + mt * 16 + r + 8 * half;
                    outp[(size_t)m * Nc + c] = acc[mt][nt][r] + bi;
                }
        }
    }
}

// -------------------------------------------- q'/k' = exp(dn*(X@pm^T)-max+eps)
// one block = 64 rows of one (b,h); 4 waves x 16 rows; F = 64 (4 n-tiles)
__global__ __launch_bounds__(128) void proj_exp_kernel(
    const __bf16* __restrict__ qk,   // [BH][4096][64]
    const float*  __restrict__ pm,   // [64][64]
    __bf16* __restrict__ outp)       // [BH][4096][64]
{
    __shared__ __align__(32) __bf16 pm_l[64 * 64];
    __shared__ float proj_l[64 * 64];

    const int tid  = threadIdx.x;
    const int lane = tid & 31, wave = tid >> 5;
    const int half = lane >> 4, mrow = lane & 15, col = lane & 15;
    const int bh    = blockIdx.x >> 6;
    const int nbase = (blockIdx.x & 63) * 64;

#pragma unroll
    for (int j = 0; j < 32; j++) {
        int idx = tid * 32 + j;
        pm_l[idx] = (__bf16)pm[idx];
    }
    __syncthreads();

    const __bf16* arow = qk + ((size_t)bh * 4096 + nbase + wave * 16 + mrow) * 64;
    v8f acc[4];
#pragma unroll
    for (int nt = 0; nt < 4; nt++) acc[nt] = v8f_zero();

#pragma unroll
    for (int ks = 0; ks < 2; ks++) {
        v16bf a;
#pragma unroll
        for (int e = 0; e < 8; e++) a[e]     = arow[ks * 32 + half * 16 + e];
#pragma unroll
        for (int e = 0; e < 8; e++) a[8 + e] = arow[ks * 32 + half * 16 + 8 + e];
#pragma unroll
        for (int nt = 0; nt < 4; nt++) {
            v16bf b = *(const v16bf*)&pm_l[(nt * 16 + col) * 64 + ks * 32 + half * 16];
            acc[nt] = __builtin_amdgcn_wmma_f32_16x16x32_bf16(
                false, a, false, b, (short)0, acc[nt], false, false);
        }
    }
    const float dn = 0.35355339059327373f;   // 64^-0.25
#pragma unroll
    for (int nt = 0; nt < 4; nt++)
#pragma unroll
        for (int r = 0; r < 8; r++)
            proj_l[(wave * 16 + r + 8 * half) * 64 + nt * 16 + col] = acc[nt][r] * dn;
    __syncthreads();

    if (tid < 64) {
        int row = tid;
        float mx = -1e30f;
#pragma unroll
        for (int f = 0; f < 64; f++) mx = fmaxf(mx, proj_l[row * 64 + f]);
        __bf16* orow = outp + ((size_t)bh * 4096 + nbase + row) * 64;
#pragma unroll
        for (int f = 0; f < 64; f++)
            orow[f] = (__bf16)__expf(proj_l[row * 64 + f] - mx + 1e-4f);
    }
}

// -------------------------------------------- kv[f,vd] = sum_n k'[n,f]*v[n,vd]
// one block per (b,h); 8 waves own 2 of the 16 (ftile,vdtile) output tiles
__global__ __launch_bounds__(256) void kv_ksum_kernel(
    const __bf16* __restrict__ kp, const __bf16* __restrict__ vp,
    float* __restrict__ kv, float* __restrict__ ksum)
{
    __shared__ __align__(32) __bf16 kT[64][32];   // [f][n]
    __shared__ __align__(32) __bf16 vT[64][32];   // [vd][n]
    __shared__ float ksp[4][64];

    const int tid  = threadIdx.x;
    const int lane = tid & 31, wave = tid >> 5;
    const int half = lane >> 4, mrow = lane & 15, col = lane & 15;
    const int bh    = blockIdx.x;
    const int ftile = wave >> 1;
    const int vd0   = (wave & 1) * 2;
    const __bf16* kb = kp + (size_t)bh * 4096 * 64;
    const __bf16* vb = vp + (size_t)bh * 4096 * 64;

    v8f acc[2];
#pragma unroll
    for (int t = 0; t < 2; t++) acc[t] = v8f_zero();

    for (int s = 0; s < 128; s++) {
        int n0 = s * 32;
#pragma unroll
        for (int j = 0; j < 8; j++) {
            int idx = tid * 8 + j;        // 0..2047 = n*64+f
            int n = idx >> 6, f = idx & 63;
            kT[f][n] = kb[(size_t)(n0 + n) * 64 + f];
            vT[f][n] = vb[(size_t)(n0 + n) * 64 + f];
        }
        __syncthreads();

        v16bf a;
        int fr = ftile * 16 + mrow;
#pragma unroll
        for (int e = 0; e < 8; e++) a[e]     = kT[fr][half * 16 + e];
#pragma unroll
        for (int e = 0; e < 8; e++) a[8 + e] = kT[fr][half * 16 + 8 + e];
#pragma unroll
        for (int t = 0; t < 2; t++) {
            v16bf b = *(const v16bf*)&vT[(vd0 + t) * 16 + col][half * 16];
            acc[t] = __builtin_amdgcn_wmma_f32_16x16x32_bf16(
                false, a, false, b, (short)0, acc[t], false, false);
        }
        __syncthreads();
    }

    float* kvo = kv + (size_t)bh * 64 * 64;
#pragma unroll
    for (int t = 0; t < 2; t++)
#pragma unroll
        for (int r = 0; r < 8; r++) {
            int f  = ftile * 16 + r + 8 * half;
            int vd = (vd0 + t) * 16 + col;
            kvo[f * 64 + vd] = acc[t][r];
        }

    // k_sum[f] = sum_n k'[n,f]
    int f = tid & 63, part = tid >> 6;
    float s = 0.f;
    for (int n = part * 1024; n < part * 1024 + 1024; n++)
        s += (float)kb[(size_t)n * 64 + f];
    ksp[part][f] = s;
    __syncthreads();
    if (tid < 64)
        ksum[bh * 64 + tid] = ksp[0][tid] + ksp[1][tid] + ksp[2][tid] + ksp[3][tid];
}

// -------------------------------------------- out = (q' @ kv) / (q'@ksum+eps)
__global__ __launch_bounds__(128) void attn_out_kernel(
    const __bf16* __restrict__ qp, const float* __restrict__ kv,
    const float* __restrict__ ksum, __bf16* __restrict__ o)
{
    __shared__ __align__(32) __bf16 kvT[64][64];   // [vd][f]
    __shared__ float ksl[64];
    __shared__ float nrm[64];

    const int tid  = threadIdx.x;
    const int lane = tid & 31, wave = tid >> 5;
    const int half = lane >> 4, mrow = lane & 15, col = lane & 15;
    const int bh    = blockIdx.x >> 6;
    const int nbase = (blockIdx.x & 63) * 64;
    const float* kvb = kv + (size_t)bh * 4096;     // 64*64 floats

#pragma unroll
    for (int j = 0; j < 32; j++) {
        int idx = tid * 32 + j;        // f*64+vd
        int f = idx >> 6, vd = idx & 63;
        kvT[vd][f] = (__bf16)kvb[idx];
    }
    if (tid < 64) ksl[tid] = ksum[bh * 64 + tid];
    __syncthreads();

    if (tid < 64) {
        int row = tid;
        const __bf16* qrow = qp + ((size_t)bh * 4096 + nbase + row) * 64;
        float s = 0.f;
#pragma unroll
        for (int f = 0; f < 64; f++) s += (float)qrow[f] * ksl[f];
        nrm[row] = 1.0f / (s + 1e-6f);
    }

    const __bf16* arow = qp + ((size_t)bh * 4096 + nbase + wave * 16 + mrow) * 64;
    v8f acc[4];
#pragma unroll
    for (int nt = 0; nt < 4; nt++) acc[nt] = v8f_zero();
#pragma unroll
    for (int ks = 0; ks < 2; ks++) {
        v16bf a;
#pragma unroll
        for (int e = 0; e < 8; e++) a[e]     = arow[ks * 32 + half * 16 + e];
#pragma unroll
        for (int e = 0; e < 8; e++) a[8 + e] = arow[ks * 32 + half * 16 + 8 + e];
#pragma unroll
        for (int nt = 0; nt < 4; nt++) {
            v16bf b = *(const v16bf*)&kvT[nt * 16 + col][ks * 32 + half * 16];
            acc[nt] = __builtin_amdgcn_wmma_f32_16x16x32_bf16(
                false, a, false, b, (short)0, acc[nt], false, false);
        }
    }
    __syncthreads();

    int b_i = bh >> 4, h = bh & 15;
#pragma unroll
    for (int nt = 0; nt < 4; nt++)
#pragma unroll
        for (int r = 0; r < 8; r++) {
            int rowl = wave * 16 + r + 8 * half;
            int nidx = nbase + rowl;
            float v = acc[nt][r] * nrm[rowl];
            o[((size_t)(b_i * 4096 + nidx)) * 1024 + h * 64 + nt * 16 + col] = (__bf16)v;
        }
}

// ---------------------------------------------------------------- launcher
extern "C" void kernel_launch(void* const* d_in, const int* in_sizes, int n_in,
                              void* d_out, int out_size, void* d_ws, size_t ws_size,
                              hipStream_t stream)
{
    const float* x     = (const float*)d_in[0];   // [4,4096,1024]
    const float* Wqkv  = (const float*)d_in[1];   // [3072,1024]
    const float* bqkv  = (const float*)d_in[2];   // [3072]
    const float* Wproj = (const float*)d_in[3];   // [1024,1024]
    const float* bproj = (const float*)d_in[4];   // [1024]
    const float* pm    = (const float*)d_in[5];   // [64,64]

    char* ws = (char*)d_ws;
    __bf16* xb   = (__bf16*)(ws);                 // 16,777,216 bf16
    __bf16* wqb  = (__bf16*)(ws + 33554432);      //  3,145,728 bf16
    __bf16* wpb  = (__bf16*)(ws + 39845888);      //  1,048,576 bf16
    __bf16* q    = (__bf16*)(ws + 41943040);      //  4,194,304 bf16 each
    __bf16* k    = q + 4194304;
    __bf16* v    = k + 4194304;
    __bf16* qp   = (__bf16*)(ws + 67108864);      //  4,194,304 bf16
    __bf16* kp   = (__bf16*)(ws + 75497472);      //  4,194,304 bf16
    float*  kvw  = (float*)(ws + 83886080);       //    262,144 f32
    float*  ksum = (float*)(ws + 84934656);       //      4,096 f32
    __bf16* o    = (__bf16*)(ws + 84951040);      // 16,777,216 bf16

    cvt_f32_bf16<<<65536, 256, 0, stream>>>(x, xb, 16777216);
    cvt_f32_bf16<<<12288, 256, 0, stream>>>(Wqkv, wqb, 3145728);
    cvt_f32_bf16<<<4096, 256, 0, stream>>>(Wproj, wpb, 1048576);

    // qkv = x @ Wqkv^T + b  -> scatter to q/k/v [B,H,N,64] bf16
    gemm_bf16_tn<0><<<dim3(128, 24), 256, 0, stream>>>(
        xb, wqb, bqkv, nullptr, 16384, 3072, 1024, q);

    proj_exp_kernel<<<4096, 128, 0, stream>>>(q, pm, qp);
    proj_exp_kernel<<<4096, 128, 0, stream>>>(k, pm, kp);

    kv_ksum_kernel<<<64, 256, 0, stream>>>(kp, v, kvw, ksum);

    attn_out_kernel<<<4096, 128, 0, stream>>>(qp, kvw, ksum, o);

    // final: out = o @ Wproj^T + b  (f32 to d_out)
    gemm_bf16_tn<1><<<dim3(128, 8), 256, 0, stream>>>(
        o, wpb, bproj, (float*)d_out, 16384, 1024, 1024, nullptr);
}
